// Model_63556926046509
// MI455X (gfx1250) — compile-verified, matrix operand order checked by reading, not run on hardware
//
#include <hip/hip_runtime.h>

// ---------------- model dims ----------------
constexpr int BB = 2, CC = 128, MMd = 4, KKm = 4, DD = 1024,
              HH = 16, AAh = 64, LLn = 4, HFF = 2048;
constexpr int EE = MMd + KKm;      // 8
constexpr int SS = CC * EE;        // 1024
constexpr float EPSf = 1e-5f;

typedef __attribute__((ext_vector_type(16))) __bf16 v16bf;
typedef __attribute__((ext_vector_type(8)))  float  v8f;
typedef __attribute__((ext_vector_type(4)))  unsigned int v4u;
typedef __attribute__((ext_vector_type(8)))  int v8i;
typedef __attribute__((ext_vector_type(4)))  int v4i;

#if defined(__has_builtin)
#if __has_builtin(__builtin_amdgcn_tensor_load_to_lds) && \
    __has_builtin(__builtin_amdgcn_s_wait_tensorcnt)
#define USE_TDM 1
#endif
#endif
#ifndef USE_TDM
#define USE_TDM 0
#endif

// ---------------- WMMA helpers ----------------
__device__ __forceinline__ v8f wmma_bf16(v16bf a, v16bf b, v8f c) {
  // D = A(16x32 bf16) * B(32x16 bf16) + C(16x16 f32)
  return __builtin_amdgcn_wmma_f32_16x16x32_bf16(false, a, false, b,
                                                 (short)0, c, false, false);
}

// A-fragment (or row-major-weight B-fragment) from f32 row-major tile
// (works for global or LDS pointers; address space inferred after inlining).
// Lane<16: row=lane, K in {0..7,16..23}; lane>=16: row=lane-16, K in {8..15,24..31}.
__device__ __forceinline__ v16bf frag_f32(const float* tile, int ld) {
  const int lane = threadIdx.x & 31;
  const int r = lane & 15, hi = lane >> 4;
  const float* p = tile + (size_t)r * ld + hi * 8;
  float4 f0 = *(const float4*)(p + 0);
  float4 f1 = *(const float4*)(p + 4);
  float4 f2 = *(const float4*)(p + 16);
  float4 f3 = *(const float4*)(p + 20);
  v16bf a;
  a[0]=(__bf16)f0.x;  a[1]=(__bf16)f0.y;  a[2]=(__bf16)f0.z;  a[3]=(__bf16)f0.w;
  a[4]=(__bf16)f1.x;  a[5]=(__bf16)f1.y;  a[6]=(__bf16)f1.z;  a[7]=(__bf16)f1.w;
  a[8]=(__bf16)f2.x;  a[9]=(__bf16)f2.y;  a[10]=(__bf16)f2.z; a[11]=(__bf16)f2.w;
  a[12]=(__bf16)f3.x; a[13]=(__bf16)f3.y; a[14]=(__bf16)f3.z; a[15]=(__bf16)f3.w;
  return a;
}

// B-fragment of a V chunk (32 keys x 16 dims), V row-major [.,64].
__device__ __forceinline__ v16bf frag_vcol(const float* __restrict__ vbase, int dcol) {
  const int lane = threadIdx.x & 31;
  const int n = lane & 15, hi = lane >> 4;
  const float* p = vbase + (size_t)hi * 8 * AAh + dcol + n;
  v16bf o;
#pragma unroll
  for (int i = 0; i < 8; ++i) o[i] = (__bf16)p[(size_t)i * AAh];
#pragma unroll
  for (int i = 0; i < 8; ++i) o[8 + i] = (__bf16)p[(size_t)(16 + i) * AAh];
  return o;
}

#if USE_TDM
// Issue a TDM 2D tile load: 32 (x, f32) by 64 (rows) tile, row stride = ldRow floats.
// D# per CDNA5 ISA ch.8: group0 {count|lds_addr|global_addr|type=2},
// group1 {data_size=4B, tensor dims, tile dims, dim0 stride}.
__device__ __forceinline__ void tdm_load_tile(const float* gsrc,
                                              unsigned lds_bytes, int ldRow) {
  unsigned long long ga = (unsigned long long)(size_t)gsrc;
  v4u g0;
  g0[0] = 1u;                                   // count=1, user mode
  g0[1] = lds_bytes;                            // lds_addr
  g0[2] = (unsigned)ga;                         // global_addr[31:0]
  g0[3] = (unsigned)((ga >> 32) & 0x01FFFFFFu) | 0x80000000u; // addr[56:32]|type=2
  v8i g1;
  g1[0] = 0x00020000;                           // workgroup_mask=0, data_size=2 (4B)
  g1[1] = (int)((unsigned)ldRow << 16);         // tensor_dim0[15:0]
  g1[2] = (int)(((unsigned)ldRow >> 16) | 0xFFFF0000u); // dim0[31:16], dim1[15:0]
  g1[3] = (int)(32u << 16);                     // dim1[31:16]=0, tile_dim0=32
  g1[4] = 64;                                   // tile_dim1=64, tile_dim2=0
  g1[5] = ldRow;                                // tensor_dim0_stride[31:0]
  g1[6] = 0;
  g1[7] = 0;
  v4i g2 = {};
  v4i g3 = {};
#if defined(__clang_major__) && (__clang_major__ >= 23)
  v8i g4 = {};
  __builtin_amdgcn_tensor_load_to_lds(g0, g1, g2, g3, g4, 0);
#else
  __builtin_amdgcn_tensor_load_to_lds(g0, g1, g2, g3, 0);
#endif
}
#endif

// ---------------- generic WMMA GEMM: C = A(MxK) * W(NxK)^T (+resid) ----------------
// grid = (Nc/64, Mr/128), block = 128 (4 waves). Each wave: 32 rows x 64 cols
// (2 A-frags x 4 B-frags = 8 WMMA per 32-K step). Weight slab staged in LDS by
// the Tensor Data Mover (double buffered), published with workgroup barriers.
template <bool RESID>
__global__ __launch_bounds__(128) void gemm_wmma(const float* __restrict__ Amat,
                                                 const float* __restrict__ Wmat,
                                                 const float* __restrict__ resid,
                                                 float* __restrict__ Cmat,
                                                 int Nc, int Kd) {
  __shared__ float wtile[2][64 * 32];
  const int tid = threadIdx.x;
  const int wave = tid >> 5, lane = tid & 31;
  const int r = lane & 15, hi = lane >> 4;
  const int bx = blockIdx.x, by = blockIdx.y;
  const int rbase = by * 128 + wave * 32;
  const float* Wblk = Wmat + (size_t)bx * 64 * Kd;
  v8f acc[2][4] = {};
#if USE_TDM
  if (wave == 0) tdm_load_tile(Wblk, (unsigned)(size_t)&wtile[0][0], Kd);
  int cur = 0;
#endif
  for (int k0 = 0; k0 < Kd; k0 += 32) {
#if USE_TDM
    if (wave == 0) __builtin_amdgcn_s_wait_tensorcnt(0);
    __syncthreads();                    // publish wtile[cur]
    if (wave == 0 && (k0 + 32) < Kd)    // prefetch next slab while computing
      tdm_load_tile(Wblk + k0 + 32, (unsigned)(size_t)&wtile[cur ^ 1][0], Kd);
    const float* wt = &wtile[cur][0];
#else
    for (int idx = tid; idx < 64 * 32; idx += 128) {
      int n = idx >> 5, kk = idx & 31;
      wtile[0][idx] = Wblk[(size_t)n * Kd + k0 + kk];
    }
    __syncthreads();
    const float* wt = &wtile[0][0];
#endif
    if (k0 + 64 < Kd)                   // stream-prefetch A (global_prefetch_b8)
      __builtin_prefetch(Amat + (size_t)(rbase + r) * Kd + k0 + 64, 0, 1);
    v16bf a0 = frag_f32(Amat + (size_t)rbase * Kd + k0, Kd);
    v16bf a1 = frag_f32(Amat + (size_t)(rbase + 16) * Kd + k0, Kd);
#pragma unroll
    for (int nt = 0; nt < 4; ++nt) {
      v16bf bfr = frag_f32(wt + nt * 16 * 32, 32);   // ds_load_b128 x4 + cvt
      acc[0][nt] = wmma_bf16(a0, bfr, acc[0][nt]);
      acc[1][nt] = wmma_bf16(a1, bfr, acc[1][nt]);
    }
#if USE_TDM
    cur ^= 1;
#else
    __syncthreads();
#endif
  }
#pragma unroll
  for (int mt = 0; mt < 2; ++mt) {
#pragma unroll
    for (int nt = 0; nt < 4; ++nt) {
#pragma unroll
      for (int j = 0; j < 8; ++j) {
        size_t row = (size_t)rbase + mt * 16 + j + 8 * hi;
        size_t idx = row * (size_t)Nc + bx * 64 + nt * 16 + r;
        if (RESID) Cmat[idx] = acc[mt][nt][j] + resid[idx];
        else       Cmat[idx] = acc[mt][nt][j];
      }
    }
  }
}

// ---------------- RMSNorm ----------------
__global__ __launch_bounds__(256) void rmsnorm_k(const float* __restrict__ x,
                                                 const float* __restrict__ w,
                                                 float* __restrict__ out) {
  __shared__ float red[256];
  const int row = blockIdx.x, tid = threadIdx.x;
  const float* xr = x + (size_t)row * DD;
  float s = 0.f;
  for (int i = tid; i < DD; i += 256) { float v = xr[i]; s += v * v; }
  red[tid] = s; __syncthreads();
  for (int off = 128; off > 0; off >>= 1) {
    if (tid < off) red[tid] += red[tid + off];
    __syncthreads();
  }
  float inv = rsqrtf(red[0] / (float)DD + EPSf);
  for (int i = tid; i < DD; i += 256) out[(size_t)row * DD + i] = xr[i] * inv * w[i];
}

// ---------------- q/k l2norm + rope, v transpose: one wave per (b,h,s) ----------------
__global__ __launch_bounds__(128) void qk_prep_k(const float* __restrict__ qkv,
                                                 float* __restrict__ qb,
                                                 float* __restrict__ kb,
                                                 float* __restrict__ vb) {
  const int gw = blockIdx.x * 4 + (threadIdx.x >> 5);
  const int lane = threadIdx.x & 31;
  const int s = gw & (SS - 1);
  const int h = (gw >> 10) & (HH - 1);
  const int b = gw >> 14;
  const size_t row = ((size_t)b * SS + s) * (size_t)(3 * DD);
  const size_t orow = ((size_t)(b * HH + h) * SS + s) * AAh;
  vb[orow + lane]      = qkv[row + 2 * DD + h * AAh + lane];
  vb[orow + lane + 32] = qkv[row + 2 * DD + h * AAh + lane + 32];
  const float ang = (float)s * __expf(-logf(10000.f) * (float)lane / 32.f);
  const float cs = cosf(ang), sn = sinf(ang);
#pragma unroll
  for (int t = 0; t < 2; ++t) {   // t=0 -> q, t=1 -> k
    const size_t off = row + (size_t)t * DD + h * AAh;
    float x1 = qkv[off + lane], x2 = qkv[off + lane + 32];
    float ss2 = x1 * x1 + x2 * x2;
#pragma unroll
    for (int d = 1; d < 32; d <<= 1) ss2 += __shfl_xor(ss2, d, 32);
    float inv = 1.f / fmaxf(sqrtf(ss2), EPSf);
    x1 *= inv; x2 *= inv;
    float* ob = (t == 0) ? qb : kb;
    ob[orow + lane]      =  x1 * cs + x2 * sn;
    ob[orow + lane + 32] = -x1 * sn + x2 * cs;
  }
}

// ---------------- flash attention: one wave per 16-query tile ----------------
__global__ __launch_bounds__(32) void flash_attn_k(const float* __restrict__ qb,
                                                   const float* __restrict__ kbuf,
                                                   const float* __restrict__ vbuf,
                                                   const int* __restrict__ doc,
                                                   float* __restrict__ attn) {
  __shared__ __bf16 ptile[16 * 32];
  const int wid = blockIdx.x;
  const int qt = wid & 63;            // S/16 = 64 tiles
  const int h  = (wid >> 6) & (HH - 1);
  const int b  = wid >> 10;
  const int lane = threadIdx.x & 31;
  const int r = lane & 15, hi = lane >> 4;
  const size_t headoff = (size_t)(b * HH + h) * SS * AAh;
  const float* Q  = qb   + headoff + (size_t)qt * 16 * AAh;
  const float* Kp = kbuf + headoff;
  const float* Vp = vbuf + headoff;

  v16bf qa0 = frag_f32(Q, AAh);
  v16bf qa1 = frag_f32(Q + 32, AAh);

  int sq[8], blkq[8], docq[8];
#pragma unroll
  for (int j = 0; j < 8; ++j) {
    sq[j]   = qt * 16 + j + 8 * hi;
    blkq[j] = sq[j] >> 3;             // / E
    docq[j] = doc[b * CC + blkq[j]];
  }
  float mrun[8], lrun[8];
#pragma unroll
  for (int j = 0; j < 8; ++j) { mrun[j] = -3e38f; lrun[j] = 0.f; }
  v8f o[4] = {};

  for (int kc = 0; kc < SS; kc += 32) {
    float sv[2][8];
#pragma unroll
    for (int sub = 0; sub < 2; ++sub) {
      const int kt = kc + sub * 16;
      v16bf kb0 = frag_f32(Kp + (size_t)kt * AAh, AAh);
      v16bf kb1 = frag_f32(Kp + (size_t)kt * AAh + 32, AAh);
      v8f s = {};
      s = wmma_bf16(qa0, kb0, s);
      s = wmma_bf16(qa1, kb1, s);
      const int sk   = kt + r;
      const int blkk = sk >> 3;
      const bool kok = (sk & 7) < KKm;
      const int dock = doc[b * CC + blkk];
#pragma unroll
      for (int j = 0; j < 8; ++j) {
        bool vis = (blkq[j] == blkk) ||
                   ((blkq[j] >= blkk) && kok && (docq[j] == dock));
        sv[sub][j] = vis ? s[j] * 0.125f : -1e30f;   // scale = 1/sqrt(64)
      }
    }
    // online softmax over this 32-key chunk
#pragma unroll
    for (int j = 0; j < 8; ++j) {
      float m = fmaxf(sv[0][j], sv[1][j]);
#pragma unroll
      for (int d = 1; d < 16; d <<= 1) m = fmaxf(m, __shfl_xor(m, d, 32));
      float nm  = fmaxf(mrun[j], m);
      float fac = __expf(mrun[j] - nm);
      mrun[j] = nm;
      lrun[j] *= fac;
#pragma unroll
      for (int t = 0; t < 4; ++t) o[t][j] *= fac;
      float p0 = __expf(sv[0][j] - nm);
      float p1 = __expf(sv[1][j] - nm);
      ptile[(j + 8 * hi) * 32 + r]      = (__bf16)p0;
      ptile[(j + 8 * hi) * 32 + 16 + r] = (__bf16)p1;
      float rs = p0 + p1;
#pragma unroll
      for (int d = 1; d < 16; d <<= 1) rs += __shfl_xor(rs, d, 32);
      lrun[j] += rs;
    }
    __syncthreads();
    // P (16x32 bf16) fragment straight from LDS
    {
      const __bf16* p = ptile + r * 32 + hi * 8;
      typedef __attribute__((ext_vector_type(8))) __bf16 v8bf;
      v8bf b0 = *(const v8bf*)(p);
      v8bf b1 = *(const v8bf*)(p + 16);
      v16bf pa = __builtin_shufflevector(b0, b1,
                   0,1,2,3,4,5,6,7,8,9,10,11,12,13,14,15);
      const float* vchunk = Vp + (size_t)kc * AAh;
#pragma unroll
      for (int t = 0; t < 4; ++t) {
        v16bf vf = frag_vcol(vchunk, t * 16);
        o[t] = wmma_bf16(pa, vf, o[t]);
      }
    }
    __syncthreads();
  }
#pragma unroll
  for (int t = 0; t < 4; ++t) {
#pragma unroll
    for (int j = 0; j < 8; ++j) {
      size_t idx = ((size_t)b * SS + sq[j]) * DD + h * AAh + t * 16 + r;
      attn[idx] = o[t][j] / lrun[j];
    }
  }
}

// ---------------- SiLU gate ----------------
__global__ __launch_bounds__(256) void gate_k(const float* __restrict__ up,
                                              float* __restrict__ g) {
  size_t idx = (size_t)blockIdx.x * 256 + threadIdx.x;   // 2048*2048 total
  size_t row = idx >> 11;
  size_t col = idx & 2047;
  float u1 = up[row * (2 * HFF) + col];
  float u2 = up[row * (2 * HFF) + HFF + col];
  g[idx] = (u1 / (1.f + __expf(-u1))) * u2;
}

// ---------------- decode permute: (b,c,M,K,V) -> (b,c,K,M*V) ----------------
__global__ __launch_bounds__(256) void permute_k(const float* __restrict__ t1,
                                                 float* __restrict__ t2) {
  int idx = blockIdx.x * 256 + threadIdx.x;  // 1024*256
  int u = idx & 255;
  int rowkc = idx >> 8;
  int k = rowkc & 3, bc = rowkc >> 2;
  int m = u >> 6, v = u & 63;
  t2[idx] = t1[((size_t)bc * 4 + m) * 256 + k * 64 + v];
}

// ---------------- build interleaved sequence ----------------
__global__ __launch_bounds__(256) void build_x_k(const float* __restrict__ xin,
                                                 const float* __restrict__ t3,
                                                 const float* __restrict__ pos,
                                                 float* __restrict__ xseq) {
  size_t idx = (size_t)blockIdx.x * 256 + threadIdx.x;   // 2*1024*1024
  int d  = (int)(idx & 1023);
  int sp = (int)((idx >> 10) & 1023);
  int b  = (int)(idx >> 20);
  int c = sp >> 3, e = sp & 7;
  float v;
  if (e < MMd) v = xin[((size_t)b * (CC * MMd) + c * MMd + e) * DD + d];
  else v = t3[(((size_t)(b * CC + c)) * KKm + (e - MMd)) * DD + d] + pos[(e - MMd) * DD + d];
  xseq[idx] = v;
}

// ---------------- output gather ----------------
__global__ __launch_bounds__(256) void gather_out_k(const float* __restrict__ xseq,
                                                    float* __restrict__ out) {
  size_t idx = (size_t)blockIdx.x * 256 + threadIdx.x;   // 256*4*1024
  int d  = (int)(idx & 1023);
  int kk = (int)((idx >> 10) & 3);
  int bc = (int)(idx >> 12);
  int b = bc >> 7, c = bc & 127;
  out[idx] = xseq[(((size_t)b * SS) + c * EE + MMd + kk) * DD + d];
}

// ---------------- host-side launch ----------------
extern "C" void kernel_launch(void* const* d_in, const int* in_sizes, int n_in,
                              void* d_out, int out_size, void* d_ws, size_t ws_size,
                              hipStream_t stream) {
  (void)in_sizes; (void)n_in; (void)out_size; (void)ws_size;
  const float* x_input = (const float*)d_in[0];
  const int*   doc     = (const int*)d_in[1];
  const float* dec_w1  = (const float*)d_in[2];
  const float* dec_w2  = (const float*)d_in[3];
  const float* pos_emb = (const float*)d_in[4];
  const float* Wqkv    = (const float*)d_in[5];
  const float* Wo      = (const float*)d_in[6];
  const float* Wup     = (const float*)d_in[7];
  const float* Wdown   = (const float*)d_in[8];
  const float* anw     = (const float*)d_in[9];
  const float* fnw     = (const float*)d_in[10];
  float* out = (float*)d_out;

  float* ws    = (float*)d_ws;
  float* xseq  = ws;                    // 2*1024*1024
  float* x1    = xseq  + 2097152;
  float* xn    = x1    + 2097152;
  float* attn  = xn    + 2097152;
  float* big   = attn  + 2097152;       // qkv(6.3M) / up(8.4M)
  float* gated = big   + 8388608;       // 4.2M
  float* qb    = gated + 4194304;       // 2.1M each
  float* kb    = qb    + 2097152;
  float* vb    = kb    + 2097152;
  float* t1    = vb    + 2097152;       // 262144
  float* t2    = t1    + 262144;        // 262144
  float* t3    = t2    + 262144;        // 1048576

  // ---- decode stage ----
  gemm_wmma<false><<<dim3(4, 8), 128, 0, stream>>>(x_input, dec_w1, nullptr, t1, 256, 1024);
  permute_k<<<1024, 256, 0, stream>>>(t1, t2);
  gemm_wmma<false><<<dim3(16, 8), 128, 0, stream>>>(t2, dec_w2, nullptr, t3, 1024, 256);
  build_x_k<<<8192, 256, 0, stream>>>(x_input, t3, pos_emb, xseq);

  // ---- transformer layers ----
  for (int li = 0; li < LLn; ++li) {
    rmsnorm_k<<<2048, 256, 0, stream>>>(xseq, anw + (size_t)li * DD, xn);
    gemm_wmma<false><<<dim3(48, 16), 128, 0, stream>>>(xn, Wqkv + (size_t)li * 3 * DD * DD,
                                                       nullptr, big, 3 * DD, DD);
    qk_prep_k<<<8192, 128, 0, stream>>>(big, qb, kb, vb);
    flash_attn_k<<<2048, 32, 0, stream>>>(qb, kb, vb, doc, attn);
    gemm_wmma<true><<<dim3(16, 16), 128, 0, stream>>>(attn, Wo + (size_t)li * DD * DD,
                                                      xseq, x1, DD, DD);
    rmsnorm_k<<<2048, 256, 0, stream>>>(x1, fnw + (size_t)li * DD, xn);
    gemm_wmma<false><<<dim3(64, 16), 128, 0, stream>>>(xn, Wup + (size_t)li * 2 * HFF * DD,
                                                       nullptr, big, 2 * HFF, DD);
    gate_k<<<16384, 256, 0, stream>>>(big, gated);
    gemm_wmma<true><<<dim3(16, 16), 128, 0, stream>>>(gated, Wdown + (size_t)li * DD * HFF,
                                                      x1, xseq, DD, HFF);
  }
  gather_out_k<<<4096, 256, 0, stream>>>(xseq, out);
}
